// MultiHeadCrossAttention_27427661152300
// MI455X (gfx1250) — matrix operand order repaired and checked by reading, not compiled, not run on hardware
//
#include <hip/hip_runtime.h>

typedef __attribute__((ext_vector_type(16))) _Float16 v16h;
typedef __attribute__((ext_vector_type(8)))  _Float16 v8h;
typedef __attribute__((ext_vector_type(8)))  float    v8f;
typedef __attribute__((ext_vector_type(4)))  float    v4f;
typedef __attribute__((ext_vector_type(4)))  unsigned int u32x4;
typedef __attribute__((ext_vector_type(8)))  int      i32x8;
typedef __attribute__((ext_vector_type(4)))  int      i32x4;

constexpr int kB  = 8;
constexpr int kC  = 512;
constexpr int kHW = 1024;   // H*W
constexpr int kS  = 1024;   // sequence length = H*W
constexpr int kNH = 8;
constexpr int kD  = 64;

// Build a 16x32-f16 WMMA fragment from two contiguous 8-half (16B) chunks.
__device__ __forceinline__ v16h frag16(const _Float16* lo, const _Float16* hi) {
  v8h l = *(const v8h*)lo;
  v8h h = *(const v8h*)hi;
  v16h r;
#pragma unroll
  for (int i = 0; i < 8; ++i) { r[i] = l[i]; r[8 + i] = h[i]; }
  return r;
}

__device__ __forceinline__ v8f wmma32(v16h a, v16h b, v8f c) {
  return __builtin_amdgcn_wmma_f32_16x16x32_f16(false, a, false, b, (short)0, c,
                                                false, false);
}

// ---------------------------------------------------------------------------
// Tensor Data Mover descriptor helpers (CDNA5 ISA ch.8, D# groups 0/1).
// 2-D tile of 4-byte elements: tile_dim0 contiguous, tile_dim1 rows of
// tensor_dim0_stride elements.
// ---------------------------------------------------------------------------
__device__ __forceinline__ u32x4 tdm_group0(unsigned lds_off,
                                            unsigned long long gaddr) {
  u32x4 g;
  g[0] = 1u;                                  // count=1, user mode, no gather
  g[1] = lds_off;                             // LDS byte address
  g[2] = (unsigned)(gaddr & 0xFFFFFFFFu);     // global_addr[31:0]
  g[3] = (unsigned)((gaddr >> 32) & 0x01FFFFFFu) | (2u << 30);  // addr[56:32] | type=2
  return g;
}

__device__ __forceinline__ i32x8 tdm_group1(unsigned td0, unsigned td1,
                                            unsigned tile0, unsigned tile1,
                                            unsigned long long s0,
                                            unsigned long long s1) {
  i32x8 g;
  g[0] = 0x20000;                             // workgroup_mask=0, data_size=4B
  g[1] = (int)((td0 & 0xFFFFu) << 16);        // abar=0 | tensor_dim0[15:0]
  g[2] = (int)((td0 >> 16) | ((td1 & 0xFFFFu) << 16));
  g[3] = (int)((td1 >> 16) | (tile0 << 16));  // tensor_dim1[31:16] | tile_dim0
  g[4] = (int)tile1;                          // tile_dim1 | tile_dim2=0
  g[5] = (int)(unsigned)(s0 & 0xFFFFFFFFu);   // tensor_dim0_stride[31:0]
  g[6] = (int)((unsigned)((s0 >> 32) & 0xFFFFu) |
               (unsigned)((s1 & 0xFFFFu) << 16));
  g[7] = (int)(unsigned)((s1 >> 16) & 0xFFFFFFFFu);
  return g;
}

__device__ __forceinline__ void tdm_load(u32x4 g0, i32x8 g1) {
  i32x4 z4 = {0, 0, 0, 0};
#if defined(__clang_major__) && __clang_major__ >= 23
  i32x8 z8 = {0, 0, 0, 0, 0, 0, 0, 0};
  __builtin_amdgcn_tensor_load_to_lds(g0, g1, z4, z4, z8, 0);
#else
  __builtin_amdgcn_tensor_load_to_lds(g0, g1, z4, z4, 0);
#endif
}

// ---------------------------------------------------------------------------
// Phase 1: 1x1-conv projection  out[o,p] = sum_i w[o,i] * x[b,i,p] + bias[o]
// Tile: 128 (o) x 128 (p) per 256-thread block, K-loop step 32.
// Tiles are DMA'd global->LDS by the Tensor Data Mover (f32), converted to
// f16 staging buffers by all waves, then consumed as WMMA fragments.
// vmode==0 -> store f16 [B,h][S][d]  (Q and K)
// vmode==1 -> store f16 [B,h][d][S]  (V, transposed for the PV B-fragments)
// ---------------------------------------------------------------------------
__global__ __launch_bounds__(256)
void qkv_proj_kernel(const float* __restrict__ x,     // [B, C, HW]
                     const float* __restrict__ w,     // [C, C]
                     const float* __restrict__ bias,  // [C]
                     _Float16* __restrict__ outp,
                     int vmode)
{
  __shared__ alignas(16) float    wStage[128][32];   // TDM dest: W tile (f32)
  __shared__ alignas(16) float    xStage[32][128];   // TDM dest: X tile (f32)
  __shared__ alignas(16) _Float16 aS[128][32];       // W tile   [o][i]  (f16)
  __shared__ alignas(16) _Float16 bS[128][32];       // X tile^T [p][i]  (f16)

  const int tid  = threadIdx.x;
  const int lane = tid & 31;
  const int wid  = tid >> 5;            // wave 0..7
  const int rl   = lane & 15;
  const int p0   = blockIdx.x * 128;
  const int m0   = blockIdx.y * 128;
  const int b    = blockIdx.z;
  const float* xb = x + (size_t)b * kC * kHW;

  const int koffA = (lane < 16) ? 0 : 8;
  const int koffB = (lane < 16) ? 0 : 8;

  // TDM descriptors (group1 is loop-invariant)
  const unsigned ldsW = (unsigned)(size_t)(&wStage[0][0]);
  const unsigned ldsX = (unsigned)(size_t)(&xStage[0][0]);
  const i32x8 g1W = tdm_group1(/*td0=*/kC,  /*td1=*/kC, /*tile0=*/32,  /*tile1=*/128,
                               /*s0=*/kC,  /*s1=*/(unsigned long long)kC * kC);
  const i32x8 g1X = tdm_group1(/*td0=*/kHW, /*td1=*/kC, /*tile0=*/128, /*tile1=*/32,
                               /*s0=*/kHW, /*s1=*/(unsigned long long)kHW * kC);
  const unsigned long long wBase = (unsigned long long)(size_t)(w + (size_t)m0 * kC);
  const unsigned long long xBase = (unsigned long long)(size_t)(xb + p0);

  v8f acc[8];
#pragma unroll
  for (int i = 0; i < 8; ++i)
    acc[i] = v8f{0.f, 0.f, 0.f, 0.f, 0.f, 0.f, 0.f, 0.f};

  for (int k0 = 0; k0 < kC; k0 += 32) {
    // --- async DMA of both f32 tiles (one wave issues; TENSORcnt tracks) ---
    if (wid == 0) {
      tdm_load(tdm_group0(ldsW, wBase + (size_t)k0 * 4), g1W);                 // W: rows o, cols k
      tdm_load(tdm_group0(ldsX, xBase + (size_t)k0 * 4 * kHW), g1X);           // X: rows k, cols p
    }
    __builtin_amdgcn_s_wait_tensorcnt(0);
    __syncthreads();

    // --- cooperative f32 -> f16 convert (X also transposed) ---
#pragma unroll
    for (int j = 0; j < 4; ++j) {
      int idx = tid * 4 + j;            // 0..1023 float4 tasks
      int r   = idx >> 3;               // row 0..127
      int c4  = idx & 7;
      v4f v = *(const v4f*)&wStage[r][c4 * 4];
      _Float16* d = &aS[r][c4 * 4];
      d[0] = (_Float16)v.x; d[1] = (_Float16)v.y;
      d[2] = (_Float16)v.z; d[3] = (_Float16)v.w;
    }
#pragma unroll
    for (int j = 0; j < 4; ++j) {
      int idx = tid * 4 + j;
      int r   = idx >> 5;               // k row 0..31
      int c4  = idx & 31;               // float4 over p
      v4f v = *(const v4f*)&xStage[r][c4 * 4];
      bS[c4 * 4 + 0][r] = (_Float16)v.x;
      bS[c4 * 4 + 1][r] = (_Float16)v.y;
      bS[c4 * 4 + 2][r] = (_Float16)v.z;
      bS[c4 * 4 + 3][r] = (_Float16)v.w;
    }
    __syncthreads();

    // --- WMMA over the K-slab ---
    const int mrow = wid * 16 + rl;
    v16h af = frag16(&aS[mrow][koffA], &aS[mrow][16 + koffA]);
#pragma unroll
    for (int ni = 0; ni < 8; ++ni) {
      const int ncol = ni * 16 + rl;
      v16h bf = frag16(&bS[ncol][koffB], &bS[ncol][koffB + 16]);
      acc[ni] = wmma32(af, bf, acc[ni]);
    }
    __syncthreads();   // protect stage buffers before next DMA / convert
  }

  // epilogue: bias, f16 convert, head-major scatter
  const int mbase = m0 + wid * 16 + ((lane < 16) ? 0 : 8);
  float bsv[8];
#pragma unroll
  for (int g = 0; g < 8; ++g) bsv[g] = bias[mbase + g];

#pragma unroll
  for (int ni = 0; ni < 8; ++ni) {
    const int s = p0 + ni * 16 + rl;
#pragma unroll
    for (int g = 0; g < 8; ++g) {
      const int o  = mbase + g;
      const int h  = o >> 6;
      const int dd = o & 63;
      float val = acc[ni][g] + bsv[g];
      size_t idx = vmode
          ? ((size_t)((b * kNH + h) * kD + dd) * kS + s)     // V: [d][S]
          : ((size_t)((b * kNH + h) * kS + s) * kD + dd);    // Q,K: [S][d]
      outp[idx] = (_Float16)val;
    }
  }
}

// ---------------------------------------------------------------------------
// Phase 2: per (b, head, 32-row s-tile): scores -> softmax -> PV
// ---------------------------------------------------------------------------
__global__ __launch_bounds__(256)
void attn_kernel(const _Float16* __restrict__ qws,   // [B,h][S][d]
                 const _Float16* __restrict__ kws,   // [B,h][S][d]
                 const _Float16* __restrict__ vws,   // [B,h][d][S]
                 float* __restrict__ out)            // (B,S,nh,d) flat
{
  __shared__ alignas(16) _Float16 pS[32][kS];        // 64 KB score/prob tile
  __shared__ float rmaxp[32][8];
  __shared__ float rsump[32][8];
  __shared__ float rinv[32];

  const int tid  = threadIdx.x;
  const int lane = tid & 31;
  const int wid  = tid >> 5;
  const int rl   = lane & 15;
  const int s0   = blockIdx.x * 32;
  const int h    = blockIdx.y;
  const int b    = blockIdx.z;

  const _Float16* Q = qws + (size_t)(b * kNH + h) * kS * kD;
  const _Float16* K = kws + (size_t)(b * kNH + h) * kS * kD;
  const _Float16* V = vws + (size_t)(b * kNH + h) * kD * kS;

  const int koffA = (lane < 16) ? 0 : 8;
  const int koffB = (lane < 16) ? 0 : 8;
  const int rhi   = (lane < 16) ? 0 : 8;

  // Q fragments for this tile's 32 rows, both 32-wide d-chunks (register resident)
  v16h aq[2][2];
#pragma unroll
  for (int mi = 0; mi < 2; ++mi) {
    const _Float16* qr = Q + (size_t)(s0 + mi * 16 + rl) * kD;
#pragma unroll
    for (int kc = 0; kc < 2; ++kc)
      aq[mi][kc] = frag16(qr + kc * 32 + koffA, qr + kc * 32 + 16 + koffA);
  }

  // ---- scores = (Q Kt) / sqrt(d), f16 into LDS; wave wid owns columns [wid*128, +128)
  const int tbase = wid * 128;
#pragma unroll
  for (int ni = 0; ni < 8; ++ni) {
    const int t = tbase + ni * 16 + rl;
    const _Float16* kr = K + (size_t)t * kD;
    v16h bf0 = frag16(kr +      koffB, kr +      koffB + 16);
    v16h bf1 = frag16(kr + 32 + koffB, kr + 32 + koffB + 16);
#pragma unroll
    for (int mi = 0; mi < 2; ++mi) {
      v8f cc = v8f{0.f, 0.f, 0.f, 0.f, 0.f, 0.f, 0.f, 0.f};
      cc = wmma32(aq[mi][0], bf0, cc);
      cc = wmma32(aq[mi][1], bf1, cc);
      const int tcol = tbase + ni * 16 + rl;
#pragma unroll
      for (int g = 0; g < 8; ++g)
        pS[mi * 16 + rhi + g][tcol] = (_Float16)(cc[g] * 0.125f);
    }
  }
  __syncthreads();

  // ---- two-pass softmax: 8 threads per row, 128 columns each
  const int r    = tid >> 3;
  const int sub  = tid & 7;
  const int cb   = sub * 128;
  float mx = -3.0e30f;
  for (int c = 0; c < 128; ++c) mx = fmaxf(mx, (float)pS[r][cb + c]);
  rmaxp[r][sub] = mx;
  __syncthreads();
  mx = rmaxp[r][0];
#pragma unroll
  for (int j = 1; j < 8; ++j) mx = fmaxf(mx, rmaxp[r][j]);
  float sm = 0.f;
  for (int c = 0; c < 128; ++c) {
    float e = __expf((float)pS[r][cb + c] - mx);
    pS[r][cb + c] = (_Float16)e;
    sm += e;
  }
  rsump[r][sub] = sm;
  __syncthreads();
  if (tid < 32) {
    float s = 0.f;
#pragma unroll
    for (int j = 0; j < 8; ++j) s += rsump[tid][j];
    rinv[tid] = 1.0f / s;
  }
  __syncthreads();

  // ---- out(32x64) = P(32x1024) x V(1024x64); wave wid owns one 16x16 subtile
  const int mi  = wid & 1;        // row stripe
  const int nci = wid >> 1;       // 16-wide d stripe
  v8f oc = v8f{0.f, 0.f, 0.f, 0.f, 0.f, 0.f, 0.f, 0.f};
  const _Float16* prow = &pS[mi * 16 + rl][0];
  const _Float16* vrow = V + (size_t)(nci * 16 + rl) * kS;
  for (int t0 = 0; t0 < kS; t0 += 32) {
    if (t0 + 256 < kS) __builtin_prefetch(vrow + t0 + 256, 0, 3);
    v16h af = frag16(prow + t0 + koffA, prow + t0 + 16 + koffA);
    v16h bf = frag16(vrow + t0 + koffB, vrow + t0 + koffB + 16);
    oc = wmma32(af, bf, oc);
  }
  const int rowb = mi * 16 + rhi;
  const int dd   = nci * 16 + rl;
#pragma unroll
  for (int g = 0; g < 8; ++g) {
    const int sl = rowb + g;
    out[((size_t)(b * kS + s0 + sl) * kNH + h) * kD + dd] = oc[g] * rinv[sl];
  }
}

// ---------------------------------------------------------------------------
extern "C" void kernel_launch(void* const* d_in, const int* in_sizes, int n_in,
                              void* d_out, int out_size, void* d_ws, size_t ws_size,
                              hipStream_t stream) {
  const float* q   = (const float*)d_in[0];
  const float* k   = (const float*)d_in[1];
  const float* v   = (const float*)d_in[2];
  const float* wq  = (const float*)d_in[3];
  const float* bq  = (const float*)d_in[4];
  const float* wk  = (const float*)d_in[5];
  const float* bk  = (const float*)d_in[6];
  const float* wvp = (const float*)d_in[7];
  const float* bv  = (const float*)d_in[8];

  const size_t elems = (size_t)kB * kNH * kS * kD;   // 4,194,304 f16 each
  _Float16* qws = (_Float16*)d_ws;
  _Float16* kws = qws + elems;
  _Float16* vws = kws + elems;

  dim3 gp(kS / 128, kC / 128, kB);   // (8, 4, 8)
  dim3 bl(256);
  qkv_proj_kernel<<<gp, bl, 0, stream>>>(q, wq,  bq, qws, 0);
  qkv_proj_kernel<<<gp, bl, 0, stream>>>(k, wk,  bk, kws, 0);
  qkv_proj_kernel<<<gp, bl, 0, stream>>>(v, wvp, bv, vws, 1);

  dim3 ga(kS / 32, kNH, kB);         // (32, 8, 8)
  attn_kernel<<<ga, bl, 0, stream>>>(qws, kws, vws, (float*)d_out);
}
